// FusedAttention_83236466197157
// MI455X (gfx1250) — matrix operand order repaired and checked by reading, not compiled, not run on hardware
//
#include <hip/hip_runtime.h>
#include <cstdint>
#include <cstddef>

// ---------------------------------------------------------------------------
// Problem constants (match reference)
// ---------------------------------------------------------------------------
constexpr int B_ = 2, S_ = 1024, H_ = 4096;
constexpr int NH_ = 32, NKV_ = 8, HD_ = 128, POOL_ = 4096;
constexpr size_t MT = (size_t)B_ * S_;          // 2048 token rows

// ---------------------------------------------------------------------------
// CDNA5 WMMA / TDM plumbing
// ---------------------------------------------------------------------------
typedef __attribute__((ext_vector_type(16))) __bf16       v16bf;
typedef __attribute__((ext_vector_type(8)))  float        v8f;
typedef __attribute__((ext_vector_type(4)))  unsigned int v4u;
typedef __attribute__((ext_vector_type(4)))  unsigned int u32x4;
typedef __attribute__((ext_vector_type(8)))  int          i32x8;
typedef __attribute__((ext_vector_type(4)))  int          i32x4;

#define WMMA_BF16(a, b, c) \
  __builtin_amdgcn_wmma_f32_16x16x32_bf16(false, (a), false, (b), (short)0, (c), false, false)

static __device__ __forceinline__ unsigned short f2bf(float x) {
  unsigned int u = __float_as_uint(x);
  u += 0x7FFFu + ((u >> 16) & 1u);              // round-to-nearest-even
  return (unsigned short)(u >> 16);
}

static __device__ __forceinline__ v8f v8f_zero() {
  v8f v;
#pragma unroll
  for (int i = 0; i < 8; ++i) v[i] = 0.0f;
  return v;
}

// A-matrix (16x32 bf16) fragment.  Lane half h holds K[16h : 16h+16) of its
// row (same order as B) -> two 16B loads.
static __device__ __forceinline__ v16bf load_frag_a(const unsigned short* rowk, int half) {
  union { v16bf v; v4u u[2]; } f;
  f.u[0] = *reinterpret_cast<const v4u*>(rowk + 16 * half);
  f.u[1] = *reinterpret_cast<const v4u*>(rowk + 16 * half + 8);
  return f.v;
}

// B-matrix (32x16 bf16) fragment.  Lane half h holds K[16h : 16h+16) of its
// column -> one 32B contiguous load (column stored contiguously in memory/LDS).
static __device__ __forceinline__ v16bf load_frag_b(const unsigned short* colk, int half) {
  union { v16bf v; v4u u[2]; } f;
  const v4u* p = reinterpret_cast<const v4u*>(colk + 16 * half);
  f.u[0] = p[0];
  f.u[1] = p[1];
  return f.v;
}

// ---------------------------------------------------------------------------
// Tensor Data Mover: async 2D bf16 tile load (global -> LDS).
// Builds D# group0/group1 per CDNA5 ISA 8.3/8.4 (count=1, type=2 "image",
// data_size=1 (2 bytes), 2D tile, groups 2/3 zero).
// This toolchain exposes the 6-arg builtin:
//   (uint32x4 g0, int32x8 g1, int32x4, int32x4, int32x8, i32 cpol)
// ---------------------------------------------------------------------------
static __device__ __forceinline__ void tdm_load_2d_bf16(
    unsigned lds_off_bytes, const void* gaddr,
    unsigned tensor_d0, unsigned tensor_d1, unsigned long long stride0_elems,
    unsigned tile_d0, unsigned tile_d1) {
  const unsigned long long ga = (unsigned long long)(uintptr_t)gaddr;
  u32x4 g0;
  g0[0] = 1u;                                            // count=1, user descriptor
  g0[1] = lds_off_bytes;                                 // lds_addr (bytes)
  g0[2] = (unsigned)(ga & 0xFFFFFFFFu);                  // global_addr[31:0]
  g0[3] = (unsigned)((ga >> 32) & 0x01FFFFFFu) | 0x80000000u;  // addr[56:32] | type=2
  i32x8 g1;
  g1[0] = 0x00010000;                                    // wg_mask=0, data_size=1 (2B)
  g1[1] = (int)((tensor_d0 & 0xFFFFu) << 16);            // tensor_dim0[15:0]
  g1[2] = (int)(((tensor_d0 >> 16) & 0xFFFFu) | ((tensor_d1 & 0xFFFFu) << 16));
  g1[3] = (int)(((tensor_d1 >> 16) & 0xFFFFu) | ((tile_d0 & 0xFFFFu) << 16));
  g1[4] = (int)(tile_d1 & 0xFFFFu);                      // tile_dim1 (tile_dim2=0)
  g1[5] = (int)(stride0_elems & 0xFFFFFFFFull);          // tensor_dim0_stride[31:0]
  g1[6] = (int)((stride0_elems >> 32) & 0xFFFFull);      // stride0[47:32], stride1=0
  g1[7] = 0;
  const i32x4 z4 = {0, 0, 0, 0};
  const i32x8 z8 = {0, 0, 0, 0, 0, 0, 0, 0};
  __builtin_amdgcn_tensor_load_to_lds(g0, g1, z4, z4, z8, 0);
}

static __device__ __forceinline__ unsigned lds_offset_of(const void* p) {
  return (unsigned)(uintptr_t)p;   // LDS maps to low 32 bits of generic address
}

// ---------------------------------------------------------------------------
// fp32 -> bf16 converters
// ---------------------------------------------------------------------------
__global__ void cvt_bf16_kernel(const float* __restrict__ src,
                                unsigned short* __restrict__ dst, int n) {
  int i = blockIdx.x * blockDim.x + threadIdx.x;
  if (i < n) dst[i] = f2bf(src[i]);
}

// src is (K x N) row-major fp32; dst is (N x K) row-major bf16.
// LDS-tiled so both the fp32 reads and bf16 writes are coalesced.
__global__ __launch_bounds__(256) void cvt_T_bf16_kernel(const float* __restrict__ src,
                                                         unsigned short* __restrict__ dst,
                                                         int K, int N) {
  __shared__ unsigned short tile[32][33];
  const int n0 = blockIdx.x * 32;
  const int k0 = blockIdx.y * 32;
  const int tx = threadIdx.x;          // 0..31
  const int ty = threadIdx.y;          // 0..7
#pragma unroll
  for (int i = 0; i < 32; i += 8)
    tile[ty + i][tx] = f2bf(src[(size_t)(k0 + ty + i) * N + n0 + tx]);
  __syncthreads();
#pragma unroll
  for (int i = 0; i < 32; i += 8)
    dst[(size_t)(n0 + ty + i) * K + k0 + tx] = tile[tx][ty + i];
}

// ---------------------------------------------------------------------------
// bf16 WMMA GEMM: C[M,N] = A[M,K] * Bt[N,K]^T, fp32 accumulate.
// One wave per block, 64x64 C tile per wave (4x4 of 16x16 WMMA tiles):
// 16 WMMAs per 8 fragment loads per K=32 step -> 2x the L2-intensity of a
// 32x32 tile.  ~220 VGPRs, fits the direct 256-VGPR window.
// ---------------------------------------------------------------------------
__global__ __launch_bounds__(32) void wmma_gemm_bf16(const unsigned short* __restrict__ A,
                                                     const unsigned short* __restrict__ Bt,
                                                     float* __restrict__ C,
                                                     int M, int N, int K) {
  const int lane = threadIdx.x;
  const int r    = lane & 15;
  const int half = lane >> 4;
  const int n0   = blockIdx.x * 64;
  const int m0   = blockIdx.y * 64;
  (void)M;

  const unsigned short* ap[4];
  const unsigned short* bp[4];
#pragma unroll
  for (int i = 0; i < 4; ++i) {
    ap[i] = A  + (size_t)(m0 + 16 * i + r) * K;
    bp[i] = Bt + (size_t)(n0 + 16 * i + r) * K;
  }

  v8f acc[4][4];
#pragma unroll
  for (int i = 0; i < 4; ++i)
#pragma unroll
    for (int j = 0; j < 4; ++j) acc[i][j] = v8f_zero();

  for (int k = 0; k < K; k += 32) {
    v16bf fa[4], fb[4];
#pragma unroll
    for (int i = 0; i < 4; ++i) fa[i] = load_frag_a(ap[i] + k, half);
#pragma unroll
    for (int j = 0; j < 4; ++j) fb[j] = load_frag_b(bp[j] + k, half);
#pragma unroll
    for (int i = 0; i < 4; ++i)
#pragma unroll
      for (int j = 0; j < 4; ++j)
        acc[i][j] = WMMA_BF16(fa[i], fb[j], acc[i][j]);
  }
#pragma unroll
  for (int i = 0; i < 4; ++i)
#pragma unroll
    for (int j = 0; j < 4; ++j)
#pragma unroll
      for (int jj = 0; jj < 8; ++jj)
        C[(size_t)(m0 + 16 * i + jj + 8 * half) * N + n0 + 16 * j + r] = acc[i][j][jj];
}

// ---------------------------------------------------------------------------
// RoPE + KV scatter.  One block per token (b,s).
// ---------------------------------------------------------------------------
__global__ void rope_scatter_kernel(const float* __restrict__ xq, const float* __restrict__ xk,
                                    const float* __restrict__ xv, const float* __restrict__ cs,
                                    const float* __restrict__ sn, const int* __restrict__ sel,
                                    unsigned short* __restrict__ qr, unsigned short* __restrict__ kc,
                                    unsigned short* __restrict__ vt, float* __restrict__ okv) {
  const int t = blockIdx.x;
  const int b = t >> 10;
  const int s = t & (S_ - 1);
  const int row = sel[t];

  for (int idx = threadIdx.x; idx < NH_ * HD_; idx += blockDim.x) {
    const int h = idx >> 7, d = idx & 127;
    const float x1 = xq[(size_t)t * H_ + idx];
    const float x2 = xq[(size_t)t * H_ + (h << 7) + ((d < 64) ? d + 64 : d - 64)];
    const float c  = cs[(size_t)s * HD_ + d];
    const float si = sn[(size_t)s * HD_ + d];
    const float v  = (d < 64) ? (x1 * c - x2 * si) : (x1 * c + x2 * si);
    qr[((size_t)((b * NH_ + h) * S_ + s)) * HD_ + d] = f2bf(v);
  }
  for (int idx = threadIdx.x; idx < NKV_ * HD_; idx += blockDim.x) {
    const int kvh = idx >> 7, d = idx & 127;
    const float x1 = xk[(size_t)t * (NKV_ * HD_) + idx];
    const float x2 = xk[(size_t)t * (NKV_ * HD_) + (kvh << 7) + ((d < 64) ? d + 64 : d - 64)];
    const float c  = cs[(size_t)s * HD_ + d];
    const float si = sn[(size_t)s * HD_ + d];
    const float kv = (d < 64) ? (x1 * c - x2 * si) : (x1 * c + x2 * si);
    kc[((size_t)((b * NKV_ + kvh) * S_ + s)) * HD_ + d] = f2bf(kv);
    okv[(size_t)row * (2 * NKV_ * HD_) + (size_t)kvh * HD_ + d] = kv;

    const float vv = xv[(size_t)t * (NKV_ * HD_) + idx];
    vt[((size_t)((b * NKV_ + kvh) * HD_ + d)) * S_ + s] = f2bf(vv);
    okv[(size_t)row * (2 * NKV_ * HD_) + (size_t)(NKV_ + kvh) * HD_ + d] = vv;
  }
}

// ---------------------------------------------------------------------------
// Causal flash attention, one wave per (b, head, 16-query block).
// K/V chunks staged into LDS by the Tensor Data Mover (TENSORcnt-tracked);
// scores + P*V via bf16 WMMA; online softmax via 16-lane shuffle reductions;
// P re-laid-out through LDS into an A fragment.
// ---------------------------------------------------------------------------
__global__ __launch_bounds__(32) void flash_attn_wmma(const unsigned short* __restrict__ Qr,
                                                      const unsigned short* __restrict__ Kc,
                                                      const unsigned short* __restrict__ Vt,
                                                      unsigned short* __restrict__ Ob) {
  __shared__ __align__(16) unsigned short sK[32 * 128];   // 32 keys x 128 d
  __shared__ __align__(16) unsigned short sV[128 * 32];   // 128 d x 32 keys
  __shared__ __align__(16) unsigned short lds_p[16 * 32];

  const int lane = threadIdx.x;
  const int r    = lane & 15;
  const int half = lane >> 4;
  const int qb = blockIdx.x, h = blockIdx.y, b = blockIdx.z;
  const int q0 = qb * 16;
  const int kvh = h >> 2;                     // n_rep = 4
  const float scale = 0.08838834764831845f;   // 1/sqrt(128)

  const unsigned short* qrow = Qr + ((size_t)((b * NH_ + h) * S_ + q0 + r)) * HD_;
  v16bf qf[4];
#pragma unroll
  for (int c = 0; c < 4; ++c) qf[c] = load_frag_a(qrow + c * 32, half);

  v8f o[8];
#pragma unroll
  for (int t = 0; t < 8; ++t) o[t] = v8f_zero();
  float mi[8], li[8];
#pragma unroll
  for (int j = 0; j < 8; ++j) { mi[j] = -1e30f; li[j] = 0.0f; }

  const unsigned short* kbase = Kc + ((size_t)(b * NKV_ + kvh) * S_) * HD_;
  const unsigned short* vbase = Vt + ((size_t)(b * NKV_ + kvh) * HD_) * S_;
  const unsigned sK_off = lds_offset_of(sK);
  const unsigned sV_off = lds_offset_of(sV);

  const int kmax = q0 + 15;
  for (int k0 = 0; k0 <= kmax; k0 += 32) {
    // ---- TDM: stage K (32x128) and V^T (128x32) chunks into LDS -----------
    asm volatile("s_wait_dscnt 0" ::: "memory");   // prior chunk's ds reads done
    tdm_load_2d_bf16(sK_off, kbase + (size_t)k0 * HD_,
                     /*tensor*/ HD_, 32, /*stride0*/ HD_, /*tile*/ HD_, 32);
    tdm_load_2d_bf16(sV_off, vbase + k0,
                     /*tensor*/ 32, HD_, /*stride0*/ S_, /*tile*/ 32, HD_);
    __builtin_amdgcn_s_wait_tensorcnt(0);

    // ---- scores: two 16-key subtiles from LDS -----------------------------
    v8f st[2];
#pragma unroll
    for (int nb = 0; nb < 2; ++nb) {
      v8f sacc = v8f_zero();
      const unsigned short* krow = sK + (nb * 16 + r) * HD_;
#pragma unroll
      for (int c = 0; c < 4; ++c) {
        v16bf kf = load_frag_b(krow + c * 32, half);
        sacc = WMMA_BF16(qf[c], kf, sacc);
      }
      st[nb] = sacc;
    }
    // ---- scale + causal mask (m = query, n = key in C layout) -------------
#pragma unroll
    for (int nb = 0; nb < 2; ++nb) {
      const int key = k0 + nb * 16 + r;
#pragma unroll
      for (int j = 0; j < 8; ++j) {
        const int q = q0 + j + 8 * half;
        const float v = st[nb][j] * scale;
        st[nb][j] = (key <= q) ? v : -1e30f;
      }
    }
    // ---- online softmax: per-row stats live across one 16-lane half -------
#pragma unroll
    for (int j = 0; j < 8; ++j) {
      float rm = fmaxf(st[0][j], st[1][j]);
      rm = fmaxf(rm, __shfl_xor(rm, 1, 32));
      rm = fmaxf(rm, __shfl_xor(rm, 2, 32));
      rm = fmaxf(rm, __shfl_xor(rm, 4, 32));
      rm = fmaxf(rm, __shfl_xor(rm, 8, 32));
      const float mn    = fmaxf(mi[j], rm);
      const float alpha = __expf(mi[j] - mn);
      const float p0 = __expf(st[0][j] - mn);
      const float p1 = __expf(st[1][j] - mn);
      st[0][j] = p0; st[1][j] = p1;
      float rs = p0 + p1;
      rs += __shfl_xor(rs, 1, 32);
      rs += __shfl_xor(rs, 2, 32);
      rs += __shfl_xor(rs, 4, 32);
      rs += __shfl_xor(rs, 8, 32);
      li[j] = li[j] * alpha + rs;
      mi[j] = mn;
#pragma unroll
      for (int t = 0; t < 8; ++t) o[t][j] *= alpha;
    }
    // ---- P: C layout (fp32) -> LDS -> A fragment (bf16) -------------------
    __syncthreads();
#pragma unroll
    for (int nb = 0; nb < 2; ++nb)
#pragma unroll
      for (int j = 0; j < 8; ++j)
        lds_p[(j + 8 * half) * 32 + nb * 16 + r] = f2bf(st[nb][j]);
    __syncthreads();
    const v16bf pf = load_frag_a(lds_p + r * 32, half);
    // ---- O += P * V  (V^T staged in LDS -> contiguous B fragments) --------
#pragma unroll
    for (int t = 0; t < 8; ++t) {
      v16bf vf = load_frag_b(sV + (t * 16 + r) * 32, half);
      o[t] = WMMA_BF16(pf, vf, o[t]);
    }
    __syncthreads();
  }

  // ---- normalize + store bf16 attn output [(b*S+s), h*128+d] --------------
#pragma unroll
  for (int t = 0; t < 8; ++t) {
#pragma unroll
    for (int j = 0; j < 8; ++j) {
      const int q = q0 + j + 8 * half;
      const float val = o[t][j] / li[j];
      Ob[((size_t)(b * S_ + q)) * H_ + h * HD_ + t * 16 + r] = f2bf(val);
    }
  }
}

// ---------------------------------------------------------------------------
// Host launcher
// ---------------------------------------------------------------------------
extern "C" void kernel_launch(void* const* d_in, const int* in_sizes, int n_in,
                              void* d_out, int out_size, void* d_ws, size_t ws_size,
                              hipStream_t stream) {
  (void)in_sizes; (void)n_in; (void)out_size; (void)ws_size;

  const float* x     = (const float*)d_in[0];
  const float* cosv  = (const float*)d_in[1];
  const float* sinv  = (const float*)d_in[2];
  const float* wq    = (const float*)d_in[3];
  const float* wk    = (const float*)d_in[4];
  const float* wv    = (const float*)d_in[5];
  const float* wo    = (const float*)d_in[6];
  const float* kvbuf = (const float*)d_in[7];
  const int*   sel   = (const int*)d_in[8];

  float* out    = (float*)d_out;
  float* out_kv = out + MT * (size_t)H_;

  char* ws = (char*)d_ws;
  size_t off = 0;
  auto take = [&](size_t bytes) -> void* {
    void* p = ws + off;
    off += (bytes + 255) & ~(size_t)255;
    return p;
  };
  unsigned short* xb  = (unsigned short*)take(MT * (size_t)H_ * 2);
  unsigned short* wqT = (unsigned short*)take((size_t)H_ * (NH_ * HD_) * 2);
  unsigned short* wkT = (unsigned short*)take((size_t)H_ * (NKV_ * HD_) * 2);
  unsigned short* wvT = (unsigned short*)take((size_t)H_ * (NKV_ * HD_) * 2);
  unsigned short* woT = (unsigned short*)take((size_t)(NH_ * HD_) * H_ * 2);
  float* xq = (float*)take(MT * (size_t)(NH_ * HD_) * 4);
  float* xk = (float*)take(MT * (size_t)(NKV_ * HD_) * 4);
  float* xv = (float*)take(MT * (size_t)(NKV_ * HD_) * 4);
  unsigned short* qr = (unsigned short*)take(MT * (size_t)(NH_ * HD_) * 2);
  unsigned short* kc = (unsigned short*)take((size_t)B_ * NKV_ * S_ * HD_ * 2);
  unsigned short* vt = (unsigned short*)take((size_t)B_ * NKV_ * S_ * HD_ * 2);
  unsigned short* ao = (unsigned short*)take(MT * (size_t)H_ * 2);

  // 1) fp32 -> bf16 conversions (weights transposed, LDS-tiled for coalescing)
  {
    int n = (int)(MT * H_);
    cvt_bf16_kernel<<<dim3((n + 255) / 256), dim3(256), 0, stream>>>(x, xb, n);
  }
  cvt_T_bf16_kernel<<<dim3((NH_ * HD_) / 32, H_ / 32), dim3(32, 8), 0, stream>>>(
      wq, wqT, H_, NH_ * HD_);
  cvt_T_bf16_kernel<<<dim3((NKV_ * HD_) / 32, H_ / 32), dim3(32, 8), 0, stream>>>(
      wk, wkT, H_, NKV_ * HD_);
  cvt_T_bf16_kernel<<<dim3((NKV_ * HD_) / 32, H_ / 32), dim3(32, 8), 0, stream>>>(
      wv, wvT, H_, NKV_ * HD_);
  cvt_T_bf16_kernel<<<dim3(H_ / 32, (NH_ * HD_) / 32), dim3(32, 8), 0, stream>>>(
      wo, woT, NH_ * HD_, H_);

  // 2) QKV projections (bf16 WMMA, fp32 out), 64x64 tile per wave
  wmma_gemm_bf16<<<dim3((NH_ * HD_) / 64, (unsigned)(MT / 64)), dim3(32), 0, stream>>>(
      xb, wqT, xq, (int)MT, NH_ * HD_, H_);
  wmma_gemm_bf16<<<dim3((NKV_ * HD_) / 64, (unsigned)(MT / 64)), dim3(32), 0, stream>>>(
      xb, wkT, xk, (int)MT, NKV_ * HD_, H_);
  wmma_gemm_bf16<<<dim3((NKV_ * HD_) / 64, (unsigned)(MT / 64)), dim3(32), 0, stream>>>(
      xb, wvT, xv, (int)MT, NKV_ * HD_, H_);

  // 3) copy input kv pool to output, then overwrite selected rows
  (void)hipMemcpyAsync(out_kv, kvbuf, (size_t)POOL_ * 2 * NKV_ * HD_ * sizeof(float),
                       hipMemcpyDeviceToDevice, stream);

  // 4) RoPE + cache writes (bf16 caches + fp32 output pool rows)
  rope_scatter_kernel<<<dim3((unsigned)MT), dim3(256), 0, stream>>>(
      xq, xk, xv, cosv, sinv, sel, qr, kc, vt, out_kv);

  // 5) causal flash attention (TDM-staged K/V chunks)
  flash_attn_wmma<<<dim3(S_ / 16, NH_, B_), dim3(32), 0, stream>>>(qr, kc, vt, ao);

  // 6) output projection straight into d_out
  wmma_gemm_bf16<<<dim3(H_ / 64, (unsigned)(MT / 64)), dim3(32), 0, stream>>>(
      ao, woT, out, (int)MT, H_, NH_ * HD_);
}
